// RNABenderLayer_55018531062586
// MI455X (gfx1250) — matrix operand order repaired and checked by reading, not compiled, not run on hardware
//
#include <hip/hip_runtime.h>
#include <hip/hip_bf16.h>
#include <math.h>

// ---------------- problem constants ----------------
#define BB   4
#define LL   2048
#define KKE  16          // edges per node
#define DD   256
#define RR   16
#define FFD  1024
#define EFD  3
#define PLU  120
#define HIDD 128
#define MROWS (BB*LL)    // 8192

typedef __attribute__((ext_vector_type(16))) _Float16 v16h;
typedef __attribute__((ext_vector_type(8)))  float    v8f;

extern __shared__ float smem_dyn[];

__device__ __forceinline__ float gelu_f(float x) {
    return 0.5f * x * (1.0f + erff(x * 0.70710678118654752f));
}

__device__ __forceinline__ void zero_acc(v8f &c) {
#pragma unroll
    for (int i = 0; i < 8; ++i) c[i] = 0.0f;
}

// ---- WMMA fragment-order helpers (16x16x32 f16) ----------------------------
// A (16x32): lane = m + 16*hi, element k' of chunk lives at slot:
//   hi = (k'>>3)&1 ; v = ((k'>>4)<<2) | ((k'&7)>>1) ; elem = 2v + (k'&1)
__device__ __forceinline__ int a_frag_pos(int m, int kc) {
    int hi = (kc >> 3) & 1;
    int v  = ((kc >> 4) << 2) | ((kc & 7) >> 1);
    return (m + (hi << 4)) * 16 + 2 * v + (kc & 1);
}
// B (32x16): lane = n + 16*hi with hi = (k'>>3)&1 ; v = ((k'>>4)<<2) | ((k'&7)>>1) ; elem = 2v+(k'&1)
__device__ __forceinline__ int b_frag_pos(int n, int kc) {
    int hi = (kc >> 3) & 1;
    int v  = ((kc >> 4) << 2) | ((kc & 7) >> 1);
    return (n + (hi << 4)) * 16 + 2 * v + (kc & 1);
}

// one 512-half chunk = v16h[32]; lane picks its vector: clean ds_load_b128 x2
__device__ __forceinline__ v16h load_a16(const _Float16 *As, int kc) {
    const v16h *p = (const v16h *)As;
    return p[kc * 32 + (threadIdx.x & 31)];
}
// packed weights in global: [ntile][kchunk][lane][16] -> global_load_b128 x2
__device__ __forceinline__ v16h load_b16(const _Float16 *Wp, int nkc, int nt, int kc) {
    const v16h *p = (const v16h *)Wp;
    return p[((size_t)nt * nkc + kc) * 32 + (threadIdx.x & 31)];
}

// pair index decode for Plucker (a<b, upper triangle of 16x16)
__device__ __forceinline__ void pl_pair(int t, int &a, int &b) {
    int aa = 0, rem = t;
    while (rem >= (RR - 1 - aa)) { rem -= (RR - 1 - aa); ++aa; }
    a = aa;
    b = aa + 1 + rem;
}

// ---------------- weight repack: (N,K) f32 -> fragment-ordered f16 ----------
__global__ void k_pack_w(const float *__restrict__ W, int N, int K,
                         _Float16 *__restrict__ Wp) {
    int Kp = (K + 31) & ~31;
    int nkc = Kp >> 5;
    int total = N * Kp;
    for (int i = blockIdx.x * 256 + threadIdx.x; i < total; i += gridDim.x * 256) {
        int n = i / Kp, k = i - n * Kp;
        float v = (k < K) ? W[(size_t)n * K + k] : 0.0f;
        Wp[((size_t)(n >> 4) * nkc + (k >> 5)) * 512 + b_frag_pos(n & 15, k & 31)] =
            (_Float16)v;
    }
}

// ---------------- LayerNorm over D=256 (one row per block, 256 threads) -----
__global__ void k_layernorm(const float *__restrict__ X, int ldx,
                            const float *__restrict__ g, const float *__restrict__ bta,
                            float *__restrict__ Y, int ldy) {
    __shared__ float red[32];
    int row = blockIdx.x;
    int t = threadIdx.x;
    float x = X[(size_t)row * ldx + t];
    float s = x, s2 = x * x;
    for (int off = 16; off > 0; off >>= 1) {
        s  += __shfl_xor(s,  off, 32);
        s2 += __shfl_xor(s2, off, 32);
    }
    int wave = t >> 5;
    if ((t & 31) == 0) { red[wave] = s; red[8 + wave] = s2; }
    __syncthreads();
    if (t == 0) {
        float a = 0.0f, b = 0.0f;
        for (int i = 0; i < 8; ++i) { a += red[i]; b += red[8 + i]; }
        float mean = a / (float)DD;
        red[16] = mean;
        red[17] = b / (float)DD - mean * mean;
    }
    __syncthreads();
    float mean = red[16], var = red[17];
    float y = (x - mean) * rsqrtf(var + 1e-5f) * g[t] + bta[t];
    Y[(size_t)row * ldy + t] = y;
}

// ---------------- generic WMMA GEMM: C = act(A @ W^T + bias) ----------------
// one block = 16-row tile; A staged to LDS in f16 fragment order; W pre-packed.
__global__ void k_gemm(const float *__restrict__ A, int lda, int Kdim,
                       const _Float16 *__restrict__ Wp, const float *__restrict__ bias,
                       int N, float *__restrict__ C, int ldc, int act) {
    _Float16 *As = (_Float16 *)smem_dyn;       // 16*Kp halves, fragment order
    int Kp = (Kdim + 31) & ~31;
    int nkc = Kp >> 5;
    int row0 = blockIdx.x * 16;
    int tid = threadIdx.x;
    for (int idx = tid; idx < 16 * Kp; idx += 256) {
        int r = idx / Kp, c = idx - r * Kp;
        float v = (c < Kdim) ? A[(size_t)(row0 + r) * lda + c] : 0.0f;
        As[(c >> 5) * 512 + a_frag_pos(r, c & 31)] = (_Float16)v;
    }
    __syncthreads();
    int wave = tid >> 5, lane = tid & 31;
    int nloc = lane & 15, hi = lane >> 4;
    for (int nt = wave; nt < (N >> 4); nt += 8) {
        v8f acc; zero_acc(acc);
#pragma unroll 2
        for (int kc = 0; kc < nkc; ++kc) {
            v16h a = load_a16(As, kc);
            v16h b = load_b16(Wp, nkc, nt, kc);
            acc = __builtin_amdgcn_wmma_f32_16x16x32_f16(false, a, false, b,
                                                         (short)0, acc, false, false);
        }
        int n = (nt << 4) + nloc;
#pragma unroll
        for (int v = 0; v < 8; ++v) {
            int m = v + (hi << 3);
            float val = acc[v] + bias[n];
            if (act == 1) val = gelu_f(val);
            C[(size_t)(row0 + m) * ldc + n] = val;
        }
    }
}

// ---------------- fused 2-layer MLP: C = GELU(A@W1^T+b1) @ W2^T + b2 (+resid)
__global__ void k_mlp(const float *__restrict__ A, int lda, int K1,
                      const _Float16 *__restrict__ W1p, const float *__restrict__ b1, int H,
                      const _Float16 *__restrict__ W2p, const float *__restrict__ b2, int N,
                      const float *__restrict__ resid, int ldr,
                      float *__restrict__ C, int ldc) {
    int K1p = (K1 + 31) & ~31;
    int nkc1 = K1p >> 5, nkc2 = H >> 5;
    _Float16 *As = (_Float16 *)smem_dyn;        // 16*K1p halves, fragment order
    _Float16 *Hs = As + 16 * K1p;               // 16*H halves, fragment order
    int row0 = blockIdx.x * 16;
    int tid = threadIdx.x;
    for (int idx = tid; idx < 16 * K1p; idx += 256) {
        int r = idx / K1p, c = idx - r * K1p;
        float v = (c < K1) ? A[(size_t)(row0 + r) * lda + c] : 0.0f;
        As[(c >> 5) * 512 + a_frag_pos(r, c & 31)] = (_Float16)v;
    }
    __syncthreads();
    int wave = tid >> 5, lane = tid & 31;
    int nloc = lane & 15, hi = lane >> 4;
    // layer 1 -> GELU -> LDS hidden tile stored directly in A-fragment order
    for (int ht = wave; ht < (H >> 4); ht += 8) {
        v8f acc; zero_acc(acc);
#pragma unroll 2
        for (int kc = 0; kc < nkc1; ++kc) {
            v16h a = load_a16(As, kc);
            v16h b = load_b16(W1p, nkc1, ht, kc);
            acc = __builtin_amdgcn_wmma_f32_16x16x32_f16(false, a, false, b,
                                                         (short)0, acc, false, false);
        }
        int n = (ht << 4) + nloc;
#pragma unroll
        for (int v = 0; v < 8; ++v) {
            int m = v + (hi << 3);
            Hs[(n >> 5) * 512 + a_frag_pos(m, n & 31)] =
                (_Float16)gelu_f(acc[v] + b1[n]);
        }
    }
    __syncthreads();
    // layer 2
    for (int nt = wave; nt < (N >> 4); nt += 8) {
        v8f acc; zero_acc(acc);
#pragma unroll 2
        for (int kc = 0; kc < nkc2; ++kc) {
            v16h a = load_a16(Hs, kc);
            v16h b = load_b16(W2p, nkc2, nt, kc);
            acc = __builtin_amdgcn_wmma_f32_16x16x32_f16(false, a, false, b,
                                                         (short)0, acc, false, false);
        }
        int n = (nt << 4) + nloc;
#pragma unroll
        for (int v = 0; v < 8; ++v) {
            int m = v + (hi << 3);
            float val = acc[v] + b2[n];
            if (resid) val += resid[(size_t)(row0 + m) * ldr + n];
            C[(size_t)(row0 + m) * ldc + n] = val;
        }
    }
}

// ---------------- backbone Plucker features (3 offsets) ---------------------
__global__ void k_plucker(const float *__restrict__ z, const unsigned char *__restrict__ mask,
                          float *__restrict__ p_all, float *__restrict__ out_p1) {
    __shared__ float zi[RR], zj[RR], praw[PLU], red[1];
    int row = blockIdx.x;
    int l = row % LL;
    int t = threadIdx.x;
    if (t < RR) zi[t] = z[(size_t)row * RR + t];
    float mi = (float)mask[row];
    const int offs[3] = {1, 2, 4};
    for (int oi = 0; oi < 3; ++oi) {
        int d = offs[oi];
        __syncthreads();
        bool inr = (l + d) < LL;
        if (t < RR) zj[t] = inr ? z[((size_t)row + d) * RR + t] : 0.0f;
        float mj = inr ? (float)mask[row + d] : 0.0f;
        __syncthreads();
        float pv = 0.0f;
        if (t < PLU) {
            int a, b; pl_pair(t, a, b);
            pv = zi[a] * zj[b] - zi[b] * zj[a];
            praw[t] = pv * pv;
        }
        __syncthreads();
        if (t == 0) {
            float ss = 0.0f;
            for (int i = 0; i < PLU; ++i) ss += praw[i];
            red[0] = 1.0f / fmaxf(sqrtf(ss), 1e-8f);
        }
        __syncthreads();
        if (t < PLU) {
            float val = pv * red[0] * mi * mj;
            p_all[(size_t)row * (3 * PLU) + oi * PLU + t] = val;
            if (oi == 0) out_p1[(size_t)row * PLU + t] = val;
        }
    }
}

// ---------------- discrete curvature kappa ----------------------------------
__global__ void k_kappa(const float *__restrict__ p1, const unsigned char *__restrict__ mask,
                        float *__restrict__ kap) {
    int row = blockIdx.x;
    int l = row % LL;
    int t = threadIdx.x;
    if (t >= PLU) return;
    float fwd = (l + 1 < LL) ? p1[((size_t)row + 1) * PLU + t] : 0.0f;
    float bwd = (l > 0)      ? p1[((size_t)row - 1) * PLU + t] : 0.0f;
    float v = (fwd - 2.0f * p1[(size_t)row * PLU + t] + bwd) * (float)mask[row];
    kap[(size_t)row * PLU + t] = v;
}

// ---------------- fused structural edge mixer (one node per block) ----------
__global__ void k_edge(const float *__restrict__ z, const float *__restrict__ hg,
                       const int *__restrict__ eidx, const float *__restrict__ efeat,
                       const _Float16 *__restrict__ bp_wp, const float *__restrict__ bp_b,
                       const float *__restrict__ as_w, const float *__restrict__ as_b,
                       float *__restrict__ out_pst, float *__restrict__ g_cat) {
    __shared__ float zi[RR];
    __shared__ float zj[KKE][RR];
    __shared__ float feat[KKE][128];                    // 123 valid cols, zero padded
    __shared__ __align__(32) _Float16 feat16[4 * 512];  // fragment order, 4 k-chunks
    __shared__ float msg[KKE][DD];
    __shared__ float ss[KKE];
    __shared__ float sc[KKE];
    __shared__ float attn[KKE];
    __shared__ int   gidx[KKE];
    __shared__ float vld[KKE];
    int row = blockIdx.x;
    int bidx = row / LL;
    int tid = threadIdx.x;
    if (tid < RR) zi[tid] = z[(size_t)row * RR + tid];
    if (tid < KKE) {
        int id = eidx[(size_t)row * KKE + tid];
        vld[tid]  = (id >= 0) ? 1.0f : 0.0f;
        gidx[tid] = (id >= 0) ? id : 0;
        ss[tid] = 0.0f;
    }
    __syncthreads();
    for (int idx = tid; idx < KKE * RR; idx += 256) {
        int e = idx >> 4, r = idx & 15;
        zj[e][r] = z[((size_t)bidx * LL + gidx[e]) * RR + r];
    }
    __syncthreads();
    for (int idx = tid; idx < KKE * PLU; idx += 256) {
        int e = idx / PLU, t = idx - e * PLU;
        int a, b; pl_pair(t, a, b);
        float pv = zi[a] * zj[e][b] - zi[b] * zj[e][a];
        feat[e][t] = pv;
        atomicAdd(&ss[e], pv * pv);
    }
    __syncthreads();
    if (tid < KKE) ss[tid] = 1.0f / fmaxf(sqrtf(ss[tid]), 1e-8f);
    __syncthreads();
    for (int idx = tid; idx < KKE * PLU; idx += 256) {
        int e = idx / PLU, t = idx - e * PLU;
        float val = feat[e][t] * ss[e] * vld[e];
        feat[e][t] = val;
        out_pst[((size_t)row * KKE + e) * PLU + t] = val;
    }
    for (int idx = tid; idx < KKE * 8; idx += 256) {
        int e = idx >> 3, j = idx & 7;
        feat[e][PLU + j] = (j < EFD) ? efeat[((size_t)row * KKE + e) * EFD + j] : 0.0f;
    }
    __syncthreads();
    // convert feature tile to f16 fragment order
    for (int idx = tid; idx < KKE * 128; idx += 256) {
        int e = idx >> 7, c = idx & 127;
        feat16[(c >> 5) * 512 + a_frag_pos(e, c & 31)] = (_Float16)feat[e][c];
    }
    __syncthreads();
    // msg = GELU(feat @ bp_w^T + bp_b) : 16(edges) x 123 -> 16 x 256 via WMMA
    int wave = tid >> 5, lane = tid & 31;
    int nloc = lane & 15, hi = lane >> 4;
    for (int nt = wave; nt < 16; nt += 8) {
        v8f acc; zero_acc(acc);
#pragma unroll
        for (int kc = 0; kc < 4; ++kc) {
            v16h a = load_a16(feat16, kc);
            v16h b = load_b16(bp_wp, 4, nt, kc);
            acc = __builtin_amdgcn_wmma_f32_16x16x32_f16(false, a, false, b,
                                                         (short)0, acc, false, false);
        }
        int n = (nt << 4) + nloc;
#pragma unroll
        for (int v = 0; v < 8; ++v) {
            int e = v + (hi << 3);
            msg[e][n] = gelu_f(acc[v] + bp_b[n]);
        }
    }
    __syncthreads();
    // scores: wave w handles edges w and w+8
    for (int e = wave; e < KKE; e += 8) {
        const float *hj = hg + ((size_t)bidx * LL + gidx[e]) * 512;
        float s = 0.0f;
        for (int d2 = lane; d2 < DD; d2 += 32)
            s += msg[e][d2] * as_w[d2] + hj[d2] * as_w[DD + d2];
        for (int off = 16; off > 0; off >>= 1) s += __shfl_xor(s, off, 32);
        if (lane == 0) {
            for (int f = 0; f < EFD; ++f)
                s += efeat[((size_t)row * KKE + e) * EFD + f] * as_w[2 * DD + f];
            s += as_b[0];
            sc[e] = (vld[e] > 0.0f) ? s : -10000.0f;
        }
    }
    __syncthreads();
    if (tid == 0) {
        float mx = -1e30f;
        for (int e = 0; e < KKE; ++e) mx = fmaxf(mx, sc[e]);
        float sum = 0.0f;
        for (int e = 0; e < KKE; ++e) { attn[e] = __expf(sc[e] - mx); sum += attn[e]; }
        float inv = 1.0f / sum;
        for (int e = 0; e < KKE; ++e) attn[e] *= inv;
    }
    __syncthreads();
    {
        float acc2 = 0.0f;
        for (int e = 0; e < KKE; ++e) acc2 += attn[e] * msg[e][tid];
        g_cat[(size_t)row * 768 + 256 + tid] = acc2;   // g_bp slot of concat
    }
}

// ---------------- gated residual: h_new = h + sigmoid(apre) * u -------------
__global__ void k_gate(const float *__restrict__ h, const float *__restrict__ u,
                       const float *__restrict__ ap, float *__restrict__ hn, int n) {
    int i = blockIdx.x * blockDim.x + threadIdx.x;
    if (i < n) {
        float a = 1.0f / (1.0f + __expf(-ap[i]));
        hn[i] = h[i] + a * u[i];
    }
}

// ---------------- host orchestration ----------------------------------------
extern "C" void kernel_launch(void *const *d_in, const int *in_sizes, int n_in,
                              void *d_out, int out_size, void *d_ws, size_t ws_size,
                              hipStream_t stream) {
    (void)in_sizes; (void)n_in; (void)out_size; (void)ws_size;
    const float *h          = (const float *)d_in[0];
    const int   *eidx       = (const int *)d_in[1];
    const float *efeat      = (const float *)d_in[2];
    const unsigned char *sm = (const unsigned char *)d_in[3];
    const float *ln1_g = (const float *)d_in[4],  *ln1_b = (const float *)d_in[5];
    const float *ln2_g = (const float *)d_in[6],  *ln2_b = (const float *)d_in[7];
    const float *Wred_w = (const float *)d_in[8], *Wred_b = (const float *)d_in[9];
    const float *bb_w1 = (const float *)d_in[10], *bb_b1 = (const float *)d_in[11];
    const float *bb_w2 = (const float *)d_in[12], *bb_b2 = (const float *)d_in[13];
    const float *cv_w1 = (const float *)d_in[14], *cv_b1 = (const float *)d_in[15];
    const float *cv_w2 = (const float *)d_in[16], *cv_b2 = (const float *)d_in[17];
    const float *bp_w  = (const float *)d_in[18], *bp_b  = (const float *)d_in[19];
    const float *as_w  = (const float *)d_in[20], *as_b  = (const float *)d_in[21];
    const float *agg_w = (const float *)d_in[22], *agg_b = (const float *)d_in[23];
    const float *Wg_w  = (const float *)d_in[24], *Wg_b  = (const float *)d_in[25];
    const float *Wa_w  = (const float *)d_in[26], *Wa_b  = (const float *)d_in[27];
    const float *ff_w1 = (const float *)d_in[28], *ff_b1 = (const float *)d_in[29];
    const float *ff_w2 = (const float *)d_in[30], *ff_b2 = (const float *)d_in[31];

    const int M = MROWS;
    float *ws    = (float *)d_ws;
    float *z     = ws;                        // M*16
    float *hg    = z     + (size_t)M * 16;    // M*512  [h_in | g]
    float *g_cat = hg    + (size_t)M * 512;   // M*768  [g_bb | g_bp | g_curv]
    float *u     = g_cat + (size_t)M * 768;   // M*256
    float *ap    = u     + (size_t)M * 256;   // M*256
    float *hn1   = ap    + (size_t)M * 256;   // M*256
    float *h2    = hn1   + (size_t)M * 256;   // M*256
    float *p_all = h2    + (size_t)M * 256;   // M*360

    // packed f16 weight regions (all sizes multiples of 512 halves -> aligned)
    _Float16 *pk     = (_Float16 *)(p_all + (size_t)M * 360);
    _Float16 *Wred_p = pk;                         // 16  x 256
    _Float16 *bb1_p  = Wred_p + 16  * 256;         // 128 x 384
    _Float16 *bb2_p  = bb1_p  + 128 * 384;         // 256 x 128
    _Float16 *cv1_p  = bb2_p  + 256 * 128;         // 128 x 128
    _Float16 *cv2_p  = cv1_p  + 128 * 128;         // 256 x 128
    _Float16 *bp_p   = cv2_p  + 256 * 128;         // 256 x 128
    _Float16 *agg_p  = bp_p   + 256 * 128;         // 256 x 768
    _Float16 *Wg_p   = agg_p  + 256 * 768;         // 256 x 512
    _Float16 *Wa_p   = Wg_p   + 256 * 512;         // 256 x 512
    _Float16 *ff1_p  = Wa_p   + 256 * 512;         // 1024 x 256
    _Float16 *ff2_p  = ff1_p  + 1024 * 256;        // 256 x 1024

    float *out    = (float *)d_out;
    float *out_h  = out;                           // M*256
    float *out_p1 = out_h  + (size_t)M * DD;       // M*120
    float *out_k  = out_p1 + (size_t)M * PLU;      // M*120
    float *out_ps = out_k  + (size_t)M * PLU;      // M*16*120

    // 0. pack all weights to f16 fragment order
    k_pack_w<<<64, 256, 0, stream>>>(Wred_w, 16,  DD,      Wred_p);
    k_pack_w<<<64, 256, 0, stream>>>(bb_w1, 128, 3 * PLU,  bb1_p);
    k_pack_w<<<64, 256, 0, stream>>>(bb_w2, 256, HIDD,     bb2_p);
    k_pack_w<<<64, 256, 0, stream>>>(cv_w1, 128, PLU,      cv1_p);
    k_pack_w<<<64, 256, 0, stream>>>(cv_w2, 256, HIDD,     cv2_p);
    k_pack_w<<<64, 256, 0, stream>>>(bp_w,  256, PLU + EFD, bp_p);
    k_pack_w<<<256, 256, 0, stream>>>(agg_w, 256, 768,     agg_p);
    k_pack_w<<<128, 256, 0, stream>>>(Wg_w,  256, 512,     Wg_p);
    k_pack_w<<<128, 256, 0, stream>>>(Wa_w,  256, 512,     Wa_p);
    k_pack_w<<<256, 256, 0, stream>>>(ff_w1, 1024, DD,     ff1_p);
    k_pack_w<<<256, 256, 0, stream>>>(ff_w2, 256, FFD,     ff2_p);

    // 1. LN1 -> hg[:, 0:256)
    k_layernorm<<<M, 256, 0, stream>>>(h, DD, ln1_g, ln1_b, hg, 512);
    // 2. z = h_in @ Wred^T + b
    k_gemm<<<M / 16, 256, 16 * 256 * sizeof(_Float16), stream>>>(
        hg, 512, DD, Wred_p, Wred_b, RR, z, RR, 0);
    // 3. backbone Plucker at offsets 1,2,4
    k_plucker<<<M, 128, 0, stream>>>(z, sm, p_all, out_p1);
    // 4. curvature
    k_kappa<<<M, 128, 0, stream>>>(out_p1, sm, out_k);
    // 5. g_bb = MLP(p_all) -> g_cat[:, 0:256)
    k_mlp<<<M / 16, 256, (16 * 384 + 16 * HIDD) * sizeof(_Float16), stream>>>(
        p_all, 3 * PLU, 3 * PLU, bb1_p, bb_b1, HIDD, bb2_p, bb_b2, DD,
        nullptr, 0, g_cat, 768);
    // 6. g_curv = MLP(kappa) -> g_cat[:, 512:768)
    k_mlp<<<M / 16, 256, (16 * 128 + 16 * HIDD) * sizeof(_Float16), stream>>>(
        out_k, PLU, PLU, cv1_p, cv_b1, HIDD, cv2_p, cv_b2, DD,
        nullptr, 0, g_cat + 512, 768);
    // 7. edge mixer -> p_struct out, g_bp -> g_cat[:, 256:512)
    k_edge<<<M, 256, 0, stream>>>(z, hg, eidx, efeat, bp_p, bp_b, as_w, as_b,
                                  out_ps, g_cat);
    // 8. g = GELU(agg(g_cat)) -> hg[:, 256:512)
    k_gemm<<<M / 16, 256, 16 * 768 * sizeof(_Float16), stream>>>(
        g_cat, 768, 768, agg_p, agg_b, DD, hg + 256, 512, 1);
    // 9/10. u = Wg(hg), apre = Wa(hg)
    k_gemm<<<M / 16, 256, 16 * 512 * sizeof(_Float16), stream>>>(
        hg, 512, 512, Wg_p, Wg_b, DD, u, DD, 0);
    k_gemm<<<M / 16, 256, 16 * 512 * sizeof(_Float16), stream>>>(
        hg, 512, 512, Wa_p, Wa_b, DD, ap, DD, 0);
    // 11. gated residual
    k_gate<<<(M * DD + 255) / 256, 256, 0, stream>>>(h, u, ap, hn1, M * DD);
    // 12. LN2
    k_layernorm<<<M, 256, 0, stream>>>(hn1, DD, ln2_g, ln2_b, h2, DD);
    // 13. FF MLP with residual -> final output
    k_mlp<<<M / 16, 256, (16 * 256 + 16 * FFD) * sizeof(_Float16), stream>>>(
        h2, DD, DD, ff1_p, ff_b1, FFD, ff2_p, ff_b2, DD,
        hn1, DD, out_h, DD);
}
